// Soft_PQ_46402826666372
// MI455X (gfx1250) — compile-verified, matrix operand order checked by reading, not run on hardware
//
#include <hip/hip_runtime.h>

typedef __attribute__((ext_vector_type(16))) _Float16 v16h;
typedef __attribute__((ext_vector_type(8)))  float    v8f;

#define NW 256      // codewords
#define NB 8        // books
#define LW 32       // dims per book
#define DT 256      // total dims
#define TAU 5.0f

// ---------------------------------------------------------------------------
// Prep: c2t[b][w] = TAU * ||C_bw||^2 ; Cf16 plain f16 copy (GEMM-1 B operand,
// contiguous along d) ; CT f16 transposed [d][w] (GEMM-2 B operand, contiguous
// along w) ; copy C -> third output region.
// ---------------------------------------------------------------------------
__global__ __launch_bounds__(256) void pq_prep(const float* __restrict__ C,
                                               float* __restrict__ c2t,
                                               _Float16* __restrict__ cf16,
                                               _Float16* __restrict__ ct,
                                               float* __restrict__ outC) {
    int i = blockIdx.x * 256 + threadIdx.x;   // 0 .. 65535
    float v = C[i];
    outC[i] = v;
    cf16[i] = (_Float16)v;
    int w = i >> 8;
    int d = i & 255;
    ct[d * NW + w] = (_Float16)v;
    if (d < NB) {
        int b = d;
        float s = 0.0f;
#pragma unroll
        for (int l = 0; l < LW; ++l) {
            float c = C[w * DT + b * LW + l];
            s += c * c;
        }
        c2t[b * NW + w] = TAU * s;
    }
}

// ---------------------------------------------------------------------------
// Main: one workgroup = 16 rows, 8 waves, wave wv owns book b = wv.
// GEMM1 (16 WMMAs): logits[16x256] = 2*tau*(x_b * C_b^T) - tau*c2
// softmax + argmax via shfl_xor within 16-lane halves (matches D layout)
// P -> LDS (f16, padded rows) -> A-layout reload
// GEMM2 (16 WMMAs): Z_b[16x32] = P * C_b
// ---------------------------------------------------------------------------
__global__ __launch_bounds__(256) void pq_main(const float* __restrict__ x,
                                               const float* __restrict__ c2t,
                                               const _Float16* __restrict__ cf16,
                                               const _Float16* __restrict__ ct,
                                               float* __restrict__ outZ,
                                               int* __restrict__ outIdx) {
    __shared__ _Float16 Psh[NB][16][264];     // 264 = 256 + pad (bank spread)

    const int tid  = threadIdx.x;
    const int wv   = tid >> 5;                // wave id == book
    const int lane = tid & 31;
    const int g    = lane >> 4;               // lane half
    const int n16  = lane & 15;
    const int b    = wv;
    const int r0   = blockIdx.x * 16;

    // ---- A operand: x tile (16 rows x 32 dims of this book), f32 -> f16 ----
    const float* xrow = x + (size_t)(r0 + n16) * DT + b * LW;
    v16h a;
#pragma unroll
    for (int i = 0; i < 8; ++i) {
        a[i]     = (_Float16)xrow[g * 8 + i];
        a[i + 8] = (_Float16)xrow[16 + g * 8 + i];
    }

    // ---- GEMM 1: xc for all 256 codewords (16 tiles of 16) ----
    v8f acc[16];
#pragma unroll
    for (int t = 0; t < 16; ++t) {
        const _Float16* crow = cf16 + (size_t)(t * 16 + n16) * DT + b * LW;
        v16h bb;
#pragma unroll
        for (int i = 0; i < 8; ++i) {
            bb[i]     = crow[g * 8 + i];
            bb[i + 8] = crow[16 + g * 8 + i];
        }
        v8f z = {};
        acc[t] = __builtin_amdgcn_wmma_f32_16x16x32_f16(
            false, a, false, bb, (short)0, z, false, false);
    }

    // ---- logits, per-lane max & argmax over t ----
    float c2v[16];
#pragma unroll
    for (int t = 0; t < 16; ++t) c2v[t] = c2t[b * NW + t * 16 + n16];

    float bv[8]; int bi[8];
#pragma unroll
    for (int r = 0; r < 8; ++r) { bv[r] = -3.4e38f; bi[r] = 0; }
#pragma unroll
    for (int t = 0; t < 16; ++t) {
#pragma unroll
        for (int r = 0; r < 8; ++r) {
            float lg = 2.0f * TAU * acc[t][r] - c2v[t];
            acc[t][r] = lg;
            if (lg > bv[r]) { bv[r] = lg; bi[r] = t * 16 + n16; }
        }
    }
    // cross-lane (within each 16-lane half: masks 1,2,4,8 keep halves apart)
#pragma unroll
    for (int m = 1; m <= 8; m <<= 1) {
#pragma unroll
        for (int r = 0; r < 8; ++r) {
            float ov = __shfl_xor(bv[r], m, 32);
            int   oi = __shfl_xor(bi[r], m, 32);
            if (ov > bv[r] || (ov == bv[r] && oi < bi[r])) { bv[r] = ov; bi[r] = oi; }
        }
    }

    // ---- exp / sum / normalize ----
    float vsum[8];
#pragma unroll
    for (int r = 0; r < 8; ++r) vsum[r] = 0.0f;
#pragma unroll
    for (int t = 0; t < 16; ++t) {
#pragma unroll
        for (int r = 0; r < 8; ++r) {
            float e = __expf(acc[t][r] - bv[r]);   // bv == row max after reduce
            acc[t][r] = e;
            vsum[r] += e;
        }
    }
#pragma unroll
    for (int m = 1; m <= 8; m <<= 1) {
#pragma unroll
        for (int r = 0; r < 8; ++r) vsum[r] += __shfl_xor(vsum[r], m, 32);
    }
    float inv[8];
#pragma unroll
    for (int r = 0; r < 8; ++r) inv[r] = 1.0f / vsum[r];

    // ---- P -> LDS (D layout: row m = r + 8*g, col w = t*16 + n16) ----
#pragma unroll
    for (int t = 0; t < 16; ++t) {
#pragma unroll
        for (int r = 0; r < 8; ++r) {
            Psh[wv][r + 8 * g][t * 16 + n16] = (_Float16)(acc[t][r] * inv[r]);
        }
    }
    __syncthreads();

    // ---- GEMM 2: Z = P * C_b  (8 K-chunks of 32, 2 column halves) ----
    v8f accZ0 = {}, accZ1 = {};
#pragma unroll
    for (int kk = 0; kk < 8; ++kk) {
        const _Float16* prow = &Psh[wv][n16][kk * 32];
        v16h ap;
#pragma unroll
        for (int i = 0; i < 8; ++i) {
            ap[i]     = prow[g * 8 + i];
            ap[i + 8] = prow[16 + g * 8 + i];
        }
        const _Float16* c0 = ct + (size_t)(b * LW + n16)      * NW + kk * 32;
        const _Float16* c1 = ct + (size_t)(b * LW + 16 + n16) * NW + kk * 32;
        v16h b0, b1;
#pragma unroll
        for (int i = 0; i < 8; ++i) {
            b0[i]     = c0[g * 8 + i];
            b0[i + 8] = c0[16 + g * 8 + i];
            b1[i]     = c1[g * 8 + i];
            b1[i + 8] = c1[16 + g * 8 + i];
        }
        accZ0 = __builtin_amdgcn_wmma_f32_16x16x32_f16(
            false, ap, false, b0, (short)0, accZ0, false, false);
        accZ1 = __builtin_amdgcn_wmma_f32_16x16x32_f16(
            false, ap, false, b1, (short)0, accZ1, false, false);
    }

    // ---- store Z (D layout -> [row][d]) ----
#pragma unroll
    for (int r = 0; r < 8; ++r) {
        size_t row = (size_t)(r0 + r + 8 * g);
        outZ[row * DT + b * LW + n16]      = accZ0[r];
        outZ[row * DT + b * LW + 16 + n16] = accZ1[r];
    }

    // ---- store idx (lanes 0 and 16 hold reduced result for their 8 rows) ----
    if (n16 == 0) {
#pragma unroll
        for (int r = 0; r < 8; ++r) {
            outIdx[(size_t)(r0 + r + 8 * g) * NB + b] = bi[r];
        }
    }
}

extern "C" void kernel_launch(void* const* d_in, const int* in_sizes, int n_in,
                              void* d_out, int out_size, void* d_ws, size_t ws_size,
                              hipStream_t stream) {
    const float* x = (const float*)d_in[0];
    const float* C = (const float*)d_in[1];
    const int nrows = in_sizes[0] / DT;      // 131072

    float* outZ   = (float*)d_out;
    int*   outIdx = (int*)(outZ + (size_t)nrows * DT);
    float* outC   = (float*)(outZ + (size_t)nrows * DT + (size_t)nrows * NB);

    char* ws = (char*)d_ws;
    float*    c2t  = (float*)ws;                              // 2048 f32
    _Float16* cf16 = (_Float16*)(ws + 8192);                  // 65536 f16
    _Float16* ct   = (_Float16*)(ws + 8192 + 131072);         // 65536 f16

    pq_prep<<<NW * DT / 256, 256, 0, stream>>>(C, c2t, cf16, ct, outC);
    pq_main<<<nrows / 16, 256, 0, stream>>>(x, c2t, cf16, ct, outZ, outIdx);
}
